// CNN_LSTMRecognizer_61306363183184
// MI455X (gfx1250) — matrix operand order repaired and checked by reading, not compiled, not run on hardware
//
#include <hip/hip_runtime.h>

// ---------------------------------------------------------------------------
// CNN + LSTM recognizer for MI455X (gfx1250, wave32, WMMA bf16).
//
// Pipeline:
//   1) conv1d(K=6)+relu+maxpool(2,1) fused, emit bf16 features [32768,1824]
//   2) bf16 weight conversion + bias fold (b_ih + b_hh)
//   3) WMMA GEMM: gates_x = feat @ w_ih^T + bias   [32768,4096] fp32
//   4) persistent LSTM kernel (32 WGs, grid barrier): per step
//        gates = gates_x[t] + h_bf16 @ w_hh^T  (WMMA, K=1024)
//        cell update (sigmoid/tanh), h_bf16 + relu(h) bf16 out
//   5) WMMA GEMM: h2 = relu(hs_relu @ h2h_w^T + b)   bf16 out
//   6) WMMA GEMM: logits = h2 @ out_w^T + b (N=48, compile-time edge) fp32
//   7) log_softmax over 48, one wave32 per row
//
// Workspace requirement: ~825 MB (dominated by gates_x 512 MB).
// ---------------------------------------------------------------------------

#define S_LEN   512
#define BATCH   64
#define D_IN    120
#define OC      16
#define KCONV   6
#define AFTER_POOL 114
#define LSTM_IN 1824           // 16 * 114
#define HID     1024
#define G4      4096           // 4 * HID
#define OUTD    48
#define SB      (S_LEN * BATCH)      // 32768

typedef __attribute__((ext_vector_type(16))) __bf16    v16bf;
typedef __attribute__((ext_vector_type(8)))  float     v8f;
typedef __attribute__((ext_vector_type(8)))  unsigned  v8u;

// ---------------- helpers ----------------

__device__ __forceinline__ unsigned short f32_bf16(float f) {
  unsigned u = __float_as_uint(f);
  unsigned r = u + 0x7FFFu + ((u >> 16) & 1u);   // round-to-nearest-even
  return (unsigned short)(r >> 16);
}

__device__ __forceinline__ float sigmoidf_fast(float x) {
  return 1.0f / (1.0f + __expf(-x));
}

// A fragment: 16x32 bf16 (MxK), row-major A with leading dim lda.
// lanes 0-15 -> M=lane, K=0..15; lanes 16-31 -> same M, K=16..31
// (contiguous order, matching load_frag_B).
__device__ __forceinline__ v16bf load_frag_A(const unsigned short* __restrict__ A,
                                             int lda, int row0, int k0, int lane) {
  const int m    = lane & 15;
  const int koff = (lane & 16) ? 16 : 0;
  const unsigned short* p = A + (size_t)(row0 + m) * lda + k0 + koff;
  v8u raw;
#pragma unroll
  for (int r = 0; r < 8; ++r) {
    const int k = 2 * r;
    raw[r] = *(const unsigned*)(p + k);
  }
  return __builtin_bit_cast(v16bf, raw);
}

// B fragment: 32x16 bf16 (KxN) where B[k][n] = W[n][k], W row-major [N,K].
// lanes 0-15 -> n=lane, K=0..15; lanes 16-31 -> n=lane-16, K=16..31.
// Unguarded variant: straight-line, 2x global_load_b128 per lane.
__device__ __forceinline__ v16bf load_frag_B(const unsigned short* __restrict__ W,
                                             int ldw, int col0, int k0, int lane) {
  const int n  = lane & 15;
  const int kb = (lane & 16) ? 16 : 0;
  const unsigned short* p = W + (size_t)(col0 + n) * ldw + k0 + kb;
  v8u raw;
#pragma unroll
  for (int r = 0; r < 8; ++r) raw[r] = *(const unsigned*)(p + 2 * r);
  return __builtin_bit_cast(v16bf, raw);
}

// Guarded variant (only for the N=48 logits GEMM edge).
__device__ __forceinline__ v16bf load_frag_B_edge(
    const unsigned short* __restrict__ W, int ldw, int col0, int k0, int lane,
    int Nlim) {
  const int n    = lane & 15;
  const int kb   = (lane & 16) ? 16 : 0;
  const int nrow = col0 + n;
  v8u raw;
  if (nrow < Nlim) {
    const unsigned short* p = W + (size_t)nrow * ldw + k0 + kb;
#pragma unroll
    for (int r = 0; r < 8; ++r) raw[r] = *(const unsigned*)(p + 2 * r);
  } else {
#pragma unroll
    for (int r = 0; r < 8; ++r) raw[r] = 0u;
  }
  return __builtin_bit_cast(v16bf, raw);
}

__device__ __forceinline__ v8f wmma_bf16(v16bf a, v16bf b, v8f c) {
  return __builtin_amdgcn_wmma_f32_16x16x32_bf16(false, a, false, b,
                                                 (short)0, c, false, false);
}

// Monotonic-counter grid barrier (persistent kernel; 32 co-resident WGs).
__device__ __forceinline__ void grid_sync(unsigned* cnt, unsigned nwg,
                                          unsigned& epoch) {
  __threadfence();
  __syncthreads();
  ++epoch;
  if (threadIdx.x == 0) {
    __hip_atomic_fetch_add(cnt, 1u, __ATOMIC_ACQ_REL, __HIP_MEMORY_SCOPE_AGENT);
    while (__hip_atomic_load(cnt, __ATOMIC_ACQUIRE, __HIP_MEMORY_SCOPE_AGENT) <
           epoch * nwg) {
      __builtin_amdgcn_s_sleep(1);
    }
  }
  __syncthreads();
}

// ---------------- small prep kernels ----------------

__global__ void __launch_bounds__(256) cvt_bf16_kernel(
    const float* __restrict__ s, unsigned short* __restrict__ d, int n) {
  int i = blockIdx.x * 256 + threadIdx.x;
  if (i < n) d[i] = f32_bf16(s[i]);
}

__global__ void __launch_bounds__(256) bias_fold_kernel(
    const float* __restrict__ a, const float* __restrict__ b,
    float* __restrict__ o, int n) {
  int i = blockIdx.x * 256 + threadIdx.x;
  if (i < n) o[i] = a[i] + b[i];
}

__global__ void __launch_bounds__(256) init_state_kernel(
    const float* __restrict__ hidden, const float* __restrict__ cell,
    unsigned short* __restrict__ h_bf, float* __restrict__ c_st,
    unsigned* __restrict__ sync_cnt) {
  int i = blockIdx.x * 256 + threadIdx.x;
  if (i == 0) sync_cnt[0] = 0u;   // ws is poisoned; reset barrier each launch
  if (i < BATCH * HID) {
    h_bf[i] = f32_bf16(hidden[i]);
    c_st[i] = cell[i];
  }
}

// conv1d(valid, K=6) + relu + maxpool(k=2, stride=1), bf16 out.
// feat layout: [sb][oc*114 + p]  == flat thread index.
__global__ void __launch_bounds__(256) conv_pool_kernel(
    const float* __restrict__ x, const float* __restrict__ w,
    const float* __restrict__ bias, unsigned short* __restrict__ feat) {
  int idx = blockIdx.x * 256 + threadIdx.x;          // < 32768*1824
  int p   = idx % AFTER_POOL;
  int oc  = (idx / AFTER_POOL) % OC;
  int sb  = idx / (AFTER_POOL * OC);
  const float* xp = x + (size_t)sb * D_IN + p;
  float a0 = bias[oc], a1 = a0;
#pragma unroll
  for (int k = 0; k < KCONV; ++k) {
    float wv = w[oc * KCONV + k];
    a0 = fmaf(xp[k], wv, a0);
    a1 = fmaf(xp[k + 1], wv, a1);
  }
  float v = fmaxf(fmaxf(a0, a1), 0.0f);              // relu(max) == max(relu)
  feat[idx] = f32_bf16(v);
}

// ---------------- tiled WMMA GEMM ----------------
// out[M,N] = act(A[M,K](bf16) @ W[N,K](bf16)^T + bias[N])
// WG = 64(M) x 128(N); 8 waves as 4x2; wave tile 16x64 (4 accumulators).
// M, K must be multiples of 64/32. N_EDGE=1 only for the logits GEMM (N=48):
// edge handling is compile-time so the hot GEMMs have straight-line K loops.
template <int ACT_RELU, int OUT_BF16, int N_EDGE>
__global__ void __launch_bounds__(256) gemm_bf16_kernel(
    const unsigned short* __restrict__ A, const unsigned short* __restrict__ W,
    const float* __restrict__ bias, void* __restrict__ out,
    int M, int N, int K) {
  const int lane = threadIdx.x & 31;
  const int wave = threadIdx.x >> 5;
  const int row0 = blockIdx.y * 64 + (wave >> 1) * 16;
  const int colb = blockIdx.x * 128 + (wave & 1) * 64;
  const int n_   = lane & 15;
  const int mb   = (lane & 16) ? 8 : 0;

  v8f acc[4];
#pragma unroll
  for (int j = 0; j < 4; ++j) {
    int c = colb + j * 16 + n_;
    float bv = (!N_EDGE || c < N) ? bias[c] : 0.0f;
#pragma unroll
    for (int r = 0; r < 8; ++r) acc[j][r] = bv;
  }

  for (int k0 = 0; k0 < K; k0 += 32) {
    if (k0 + 32 < K)   // light L2 prefetch of next A block
      __builtin_prefetch(A + (size_t)(row0 + n_) * K + k0 + 32, 0, 1);
    v16bf a = load_frag_A(A, K, row0, k0, lane);
#pragma unroll
    for (int j = 0; j < 4; ++j) {
      v16bf b = N_EDGE ? load_frag_B_edge(W, K, colb + j * 16, k0, lane, N)
                       : load_frag_B(W, K, colb + j * 16, k0, lane);
      acc[j] = wmma_bf16(a, b, acc[j]);
    }
  }

#pragma unroll
  for (int j = 0; j < 4; ++j) {
    int c = colb + j * 16 + n_;
    if (!N_EDGE || c < N) {
#pragma unroll
      for (int r = 0; r < 8; ++r) {
        float v = acc[j][r];
        if (ACT_RELU) v = fmaxf(v, 0.0f);
        size_t o = (size_t)(row0 + mb + r) * N + c;
        if (OUT_BF16) ((unsigned short*)out)[o] = f32_bf16(v);
        else          ((float*)out)[o] = v;
      }
    }
  }
}

// ---------------- persistent LSTM ----------------
// grid = 32 WGs x 256 threads. Per step:
//   phase 1: gate_buf[64,4096] = gates_x[t] + h_bf @ w_hh^T  (WMMA)
//   phase 2: elementwise cell update; write h_bf (recurrence) and
//            hs_relu[t] = relu(h) (bf16, feeds h2h GEMM)
__global__ void __launch_bounds__(256) lstm_persistent_kernel(
    const float* __restrict__ gates_x, const unsigned short* __restrict__ whh,
    unsigned short* __restrict__ h_bf, float* __restrict__ c_st,
    float* __restrict__ gate_buf, unsigned short* __restrict__ hs_relu,
    unsigned* __restrict__ sync_cnt, int steps) {
  const int lane = threadIdx.x & 31;
  const int wave = threadIdx.x >> 5;
  const int row0 = (wave >> 1) * 16;                       // M strip (batch)
  const int colb = (int)blockIdx.x * 128 + (wave & 1) * 64;
  const int n_   = lane & 15;
  const int mb   = (lane & 16) ? 8 : 0;
  const unsigned nwg = gridDim.x;
  unsigned epoch = 0;

  for (int t = 0; t < steps; ++t) {
    // ---- phase 1: recurrent GEMM tile, C initialized from gates_x[t] ----
    const float* gx = gates_x + (size_t)t * BATCH * G4;
    v8f acc[4];
#pragma unroll
    for (int j = 0; j < 4; ++j)
#pragma unroll
      for (int r = 0; r < 8; ++r)
        acc[j][r] = gx[(size_t)(row0 + mb + r) * G4 + colb + j * 16 + n_];

    for (int k0 = 0; k0 < HID; k0 += 32) {
      v16bf a = load_frag_A(h_bf, HID, row0, k0, lane);
#pragma unroll
      for (int j = 0; j < 4; ++j) {
        v16bf b = load_frag_B(whh, HID, colb + j * 16, k0, lane);
        acc[j] = wmma_bf16(a, b, acc[j]);
      }
    }
#pragma unroll
    for (int j = 0; j < 4; ++j)
#pragma unroll
      for (int r = 0; r < 8; ++r)
        gate_buf[(size_t)(row0 + mb + r) * G4 + colb + j * 16 + n_] = acc[j][r];

    grid_sync(sync_cnt, nwg, epoch);

    // ---- phase 2: cell update (torch gate order i,f,g,o) ----
    const int base = (int)blockIdx.x * 2048;               // 65536 / 32 WGs
#pragma unroll
    for (int i = 0; i < 8; ++i) {
      int idx = base + (int)threadIdx.x + i * 256;         // b*1024 + h
      int b   = idx >> 10;
      int h   = idx & (HID - 1);
      const float* gb = gate_buf + (size_t)b * G4;
      float ig = sigmoidf_fast(gb[h]);
      float fg = sigmoidf_fast(gb[HID + h]);
      float gg = tanhf(gb[2 * HID + h]);
      float og = sigmoidf_fast(gb[3 * HID + h]);
      float c  = fg * c_st[idx] + ig * gg;
      float hv = og * tanhf(c);
      c_st[idx] = c;
      h_bf[idx] = f32_bf16(hv);
      hs_relu[(size_t)t * BATCH * HID + idx] = f32_bf16(fmaxf(hv, 0.0f));
    }

    grid_sync(sync_cnt, nwg, epoch);
  }
}

// ---------------- log-softmax over 48, one wave32 per row ----------------
__global__ void __launch_bounds__(256) logsoftmax48_kernel(
    const float* __restrict__ logits, float* __restrict__ out, int rows) {
  const int lane = threadIdx.x & 31;
  const int row  = blockIdx.x * 8 + (threadIdx.x >> 5);
  if (row >= rows) return;
  const float* x = logits + (size_t)row * OUTD;
  float a = x[lane];
  float b = (lane < 16) ? x[32 + lane] : -3.4e38f;
  float m = fmaxf(a, b);
#pragma unroll
  for (int off = 16; off > 0; off >>= 1) m = fmaxf(m, __shfl_xor(m, off, 32));
  float s = __expf(a - m) + ((lane < 16) ? __expf(b - m) : 0.0f);
#pragma unroll
  for (int off = 16; off > 0; off >>= 1) s += __shfl_xor(s, off, 32);
  float lse = m + __logf(s);
  float* o = out + (size_t)row * OUTD;
  o[lane] = a - lse;
  if (lane < 16) o[32 + lane] = b - lse;
}

// ---------------- host launcher ----------------

extern "C" void kernel_launch(void* const* d_in, const int* in_sizes, int n_in,
                              void* d_out, int out_size, void* d_ws,
                              size_t ws_size, hipStream_t stream) {
  (void)in_sizes; (void)n_in; (void)out_size; (void)ws_size;

  const float* input_ = (const float*)d_in[0];
  const float* hidden = (const float*)d_in[1];
  const float* cell   = (const float*)d_in[2];
  const float* conv_w = (const float*)d_in[3];
  const float* conv_b = (const float*)d_in[4];
  const float* w_ih   = (const float*)d_in[5];
  const float* w_hh   = (const float*)d_in[6];
  const float* b_ih   = (const float*)d_in[7];
  const float* b_hh   = (const float*)d_in[8];
  const float* h2h_w  = (const float*)d_in[9];
  const float* h2h_b  = (const float*)d_in[10];
  const float* out_w  = (const float*)d_in[11];
  const float* out_b  = (const float*)d_in[12];
  float* out = (float*)d_out;

  char* ws = (char*)d_ws;
  size_t off = 0;
  auto take = [&](size_t bytes) -> char* {
    char* p = ws + off;
    off += (bytes + 255) & ~(size_t)255;
    return p;
  };

  unsigned short* feat_bf  = (unsigned short*)take((size_t)SB * LSTM_IN * 2);
  unsigned short* wih_bf   = (unsigned short*)take((size_t)G4 * LSTM_IN * 2);
  unsigned short* whh_bf   = (unsigned short*)take((size_t)G4 * HID * 2);
  unsigned short* h2hw_bf  = (unsigned short*)take((size_t)HID * HID * 2);
  unsigned short* outw_bf  = (unsigned short*)take((size_t)OUTD * HID * 2);
  float*          bias4    = (float*)take((size_t)G4 * 4);
  float*          gates_x  = (float*)take((size_t)SB * G4 * 4);    // 512 MB
  float*          gate_buf = (float*)take((size_t)BATCH * G4 * 4);
  float*          c_st     = (float*)take((size_t)BATCH * HID * 4);
  unsigned short* h_bf     = (unsigned short*)take((size_t)BATCH * HID * 2);
  unsigned short* hs_relu  = (unsigned short*)take((size_t)SB * HID * 2);
  unsigned short* h2_bf    = (unsigned short*)take((size_t)SB * HID * 2);
  float*          logits   = (float*)take((size_t)SB * OUTD * 4);
  unsigned*       sync_cnt = (unsigned*)take(256);

  // 1) state / barrier init
  init_state_kernel<<<(BATCH * HID + 255) / 256, 256, 0, stream>>>(
      hidden, cell, h_bf, c_st, sync_cnt);

  // 2) weight conversions + bias fold
  cvt_bf16_kernel<<<(G4 * LSTM_IN + 255) / 256, 256, 0, stream>>>(
      w_ih, wih_bf, G4 * LSTM_IN);
  cvt_bf16_kernel<<<(G4 * HID + 255) / 256, 256, 0, stream>>>(
      w_hh, whh_bf, G4 * HID);
  cvt_bf16_kernel<<<(HID * HID + 255) / 256, 256, 0, stream>>>(
      h2h_w, h2hw_bf, HID * HID);
  cvt_bf16_kernel<<<(OUTD * HID + 255) / 256, 256, 0, stream>>>(
      out_w, outw_bf, OUTD * HID);
  bias_fold_kernel<<<(G4 + 255) / 256, 256, 0, stream>>>(b_ih, b_hh, bias4, G4);

  // 3) conv + relu + pool -> bf16 features
  conv_pool_kernel<<<(SB * LSTM_IN) / 256, 256, 0, stream>>>(
      input_, conv_w, conv_b, feat_bf);

  // 4) gates_x = feat @ w_ih^T + (b_ih + b_hh)
  gemm_bf16_kernel<0, 0, 0><<<dim3(G4 / 128, SB / 64), 256, 0, stream>>>(
      feat_bf, wih_bf, bias4, gates_x, SB, G4, LSTM_IN);

  // 5) persistent LSTM over 512 steps
  lstm_persistent_kernel<<<G4 / 128, 256, 0, stream>>>(
      gates_x, whh_bf, h_bf, c_st, gate_buf, hs_relu, sync_cnt, S_LEN);

  // 6) h2 = relu(hs_relu @ h2h_w^T + h2h_b), bf16 out
  gemm_bf16_kernel<1, 1, 0><<<dim3(HID / 128, SB / 64), 256, 0, stream>>>(
      hs_relu, h2hw_bf, h2h_b, h2_bf, SB, HID, HID);

  // 7) logits = h2 @ out_w^T + out_b   (N=48, compile-time edge handling)
  gemm_bf16_kernel<0, 0, 1><<<dim3(1, SB / 64), 256, 0, stream>>>(
      h2_bf, outw_bf, out_b, logits, SB, OUTD, HID);

  // 8) log-softmax
  logsoftmax48_kernel<<<SB / 8, 256, 0, stream>>>(logits, out, SB);
}
